// IBRNetInnerField_13048110645608
// MI455X (gfx1250) — compile-verified, matrix operand order checked by reading, not run on hardware
//
#include <hip/hip_runtime.h>
#include <hip/hip_bf16.h>
#include <math.h>

typedef __attribute__((ext_vector_type(16))) _Float16 v16h;
typedef __attribute__((ext_vector_type(8)))  float    v8f;
typedef __attribute__((ext_vector_type(4)))  float    v4f;
typedef _Float16 f16;

#define RR 256
#define SS 64
#define VW 8
#define CH 128
#define NROWS_VIEW (RR*VW*SS)   /* 131072 */
#define NROWS_RAY  (RR*SS)      /* 16384  */

union Frag { v16h h; v4f f[2]; };

__device__ __forceinline__ void load_frag(Frag& fr, const f16* __restrict__ p) {
  // VGPR0-3 <- K = khalf*8 .. +7 ; VGPR4-7 <- K = 16+khalf*8 .. +7 (ISA 16-bit A 16x32 layout)
  fr.f[0] = *reinterpret_cast<const v4f*>(p);
  fr.f[1] = *reinterpret_cast<const v4f*>(p + 16);
}

// ---------------------------------------------------------------------------
// Templated GEMM core: JT = number of valid 16-wide n-tiles for this wave.
// All control flow in here is either compile-time (JT) or scalar-uniform, so
// EXEC stays all-ones around every v_wmma.
// ---------------------------------------------------------------------------
template<int JT>
__device__ __forceinline__ void gemm_core(
    const f16* __restrict__ A, int lda,
    const f16* __restrict__ Bt, int ldb,
    const float* __restrict__ bias,
    const float* __restrict__ cinit, int cinit_mode,
    int act,
    f16* __restrict__ outH, float* __restrict__ outF, int ldo,
    int N, int K, int mt, int nt0)
{
  const int lane  = threadIdx.x & 31;
  const int khalf = lane >> 4;
  const int lrow  = lane & 15;

  const f16* aRow = A + (size_t)(mt * 16 + lrow) * lda + khalf * 8;
  const f16* bRow[JT];
  #pragma unroll
  for (int j = 0; j < JT; ++j)
    bRow[j] = Bt + (size_t)((nt0 + j) * 16 + lrow) * ldb + khalf * 8;

  v8f acc[JT] = {};
  const int kTiles = K >> 5;
  for (int kt = 0; kt < kTiles; ++kt) {
    Frag a;
    load_frag(a, aRow + kt * 32);
    if (kt + 1 < kTiles) __builtin_prefetch(aRow + (kt + 1) * 32, 0, 3);
    #pragma unroll
    for (int j = 0; j < JT; ++j) {
      Frag b;
      load_frag(b, bRow[j] + kt * 32);
      acc[j] = __builtin_amdgcn_wmma_f32_16x16x32_f16(
          false, a.h, false, b.h, (short)0, acc[j], false, false);
    }
  }

  // Epilogue. C/D layout: VGPR i -> m = khalf*8 + i (within tile), n = lane&15.
  #pragma unroll
  for (int j = 0; j < JT; ++j) {
    int n = (nt0 + j) * 16 + lrow;
    bool nOk = (n < N);
    float bv = (bias != nullptr && nOk) ? bias[n] : 0.0f;
    #pragma unroll
    for (int i = 0; i < 8; ++i) {
      int m = mt * 16 + khalf * 8 + i;
      float v = acc[j][i] + bv;
      if (cinit_mode == 1) {
        int sr = ((m >> 9) << 6) + (m & 63);     // ray*64 + sample
        v += cinit[(size_t)sr * 256 + n];
      }
      if (act == 1) v = fmaxf(v, 0.0f);
      if (!nOk) continue;
      if (act == 2) {
        // d_out = [density 16384 | rgb 16384*3 | features 16384*128]
        if (n == 0) {
          outF[m] = (v > 20.0f) ? v : log1pf(expf(v));                          // softplus
        } else if (n < 4) {
          outF[NROWS_RAY + (size_t)m * 3 + (n - 1)] = 1.0f / (1.0f + expf(-v)); // sigmoid
        } else if (n < 132) {
          outF[(size_t)NROWS_RAY * 4 + (size_t)m * 128 + (n - 4)] = v;
        }
      } else {
        if (outH) outH[(size_t)m * ldo + n] = (f16)v;
        if (outF) outF[(size_t)m * ldo + n] = v;
      }
    }
  }
}

// ---------------------------------------------------------------------------
// Generic WMMA GEMM: out = act(A[M][K](f16) @ Bt[N][K]^T(f16) + bias + cinit)
// Each wave computes a 16 x (16*JT) tile; wave id forced into an SGPR so all
// tile-selection control flow is scalar (no EXEC masking around WMMA).
// ---------------------------------------------------------------------------
__global__ __launch_bounds__(256) void wmma_gemm(
    const f16* __restrict__ A, int lda,
    const f16* __restrict__ Bt, int ldb,
    const float* __restrict__ bias,
    const float* __restrict__ cinit, int cinit_mode,
    int act,
    f16* __restrict__ outH, float* __restrict__ outF, int ldo,
    int N, int K, int groups, int totalWaves)
{
  int wid = __builtin_amdgcn_readfirstlane(
      (int)(blockIdx.x * 8 + (threadIdx.x >> 5)));
  if (wid >= totalWaves) return;                 // scalar branch
  int ntiles = (N + 15) >> 4;
  int mt  = wid / groups;
  int g   = wid % groups;
  int nt0 = g * 4;
  int nvalid = ntiles - nt0;
  if (nvalid >= 4) {
    gemm_core<4>(A, lda, Bt, ldb, bias, cinit, cinit_mode, act, outH, outF, ldo, N, K, mt, nt0);
  } else if (nvalid == 3) {
    gemm_core<3>(A, lda, Bt, ldb, bias, cinit, cinit_mode, act, outH, outF, ldo, N, K, mt, nt0);
  } else if (nvalid == 2) {
    gemm_core<2>(A, lda, Bt, ldb, bias, cinit, cinit_mode, act, outH, outF, ldo, N, K, mt, nt0);
  } else {
    gemm_core<1>(A, lda, Bt, ldb, bias, cinit, cinit_mode, act, outH, outF, ldo, N, K, mt, nt0);
  }
}

// ---------------------------------------------------------------------------
// Weight prep: Bt[n][dstK0+kk] = W[srcRow0+kk][n] (f32->f16), zero pad.
// ---------------------------------------------------------------------------
__global__ void prep_bt(const float* __restrict__ W, int ldw, int srcRow0, int nrows,
                        f16* __restrict__ Bt, int ldbt, int dstK0, int dstKcount,
                        int Npad, int Nsrc)
{
  int tid = blockIdx.x * blockDim.x + threadIdx.x;
  int total = Npad * dstKcount;
  if (tid >= total) return;
  int n  = tid / dstKcount;
  int kk = tid - n * dstKcount;
  float v = 0.0f;
  if (kk < nrows && n < Nsrc) v = W[(size_t)(srcRow0 + kk) * ldw + n];
  Bt[(size_t)n * ldbt + dstK0 + kk] = (f16)v;
}

// ---------------------------------------------------------------------------
// Projection + bilinear gather: feats[(b*64+s)][c], b = ray*8+view.
// One block per sample point, 128 threads over channels.
// ---------------------------------------------------------------------------
__global__ __launch_bounds__(128) void project_gather(
    const float* __restrict__ positions, const float* __restrict__ w2c,
    const float* __restrict__ fx, const float* __restrict__ fy,
    const float* __restrict__ cx, const float* __restrict__ cy,
    const float* __restrict__ imgf, const float* __restrict__ uvs,
    f16* __restrict__ feats)
{
  int pid = blockIdx.x;          // b*64 + s
  int b = pid >> 6, s = pid & 63;
  int r = b >> 3;
  int c = threadIdx.x;

  const float* P = positions + (size_t)(r * 64 + s) * 3;
  float px = P[0], py = P[1], pz = P[2];
  const float* Mw = w2c + (size_t)b * 16;
  float xn = Mw[0]*px + Mw[1]*py + Mw[2]*pz  + Mw[3];
  float yn = Mw[4]*px + Mw[5]*py + Mw[6]*pz  + Mw[7];
  float zn = Mw[8]*px + Mw[9]*py + Mw[10]*pz + Mw[11];
  float invz = 1.0f / zn;
  float u  = (-xn * invz) *  fx[b] + cx[b];
  float vv = (-yn * invz) * (-fy[b]) + cy[b];
  float x = u  * uvs[0];
  float y = vv * uvs[1];
  x = fminf(fmaxf(x, 0.0f), 31.0f);
  y = fminf(fmaxf(y, 0.0f), 31.0f);
  float x0f = floorf(x), y0f = floorf(y);
  int x0 = (int)x0f, y0 = (int)y0f;
  int x1 = (x0 + 1 > 31) ? 31 : x0 + 1;
  int y1 = (y0 + 1 > 31) ? 31 : y0 + 1;
  float wx = x - x0f, wy = y - y0f;

  const float* F = imgf + ((size_t)b * CH + c) * 1024;   // 32*32 plane
  float f00 = F[y0 * 32 + x0], f01 = F[y0 * 32 + x1];
  float f10 = F[y1 * 32 + x0], f11 = F[y1 * 32 + x1];
  float o = f00 * (1.0f - wx) * (1.0f - wy) + f01 * wx * (1.0f - wy)
          + f10 * (1.0f - wx) * wy          + f11 * wx * wy;
  feats[(size_t)pid * CH + c] = (f16)o;
}

// ---------------------------------------------------------------------------
// Shared input for the fp MLP: [mean(128) | var(128,ddof=1) | nerf_enc(39 pad 64)]
// ---------------------------------------------------------------------------
__global__ __launch_bounds__(128) void build_shared(
    const f16* __restrict__ feats, const float* __restrict__ positions,
    f16* __restrict__ shin)
{
  int sr = blockIdx.x;           // ray*64 + samp
  int r = sr >> 6, s = sr & 63;
  int c = threadIdx.x;

  float v[VW]; float mean = 0.0f;
  #pragma unroll
  for (int vi = 0; vi < VW; ++vi) {
    v[vi] = (float)feats[(size_t)(((r * 8 + vi) * 64) + s) * CH + c];
    mean += v[vi];
  }
  mean *= 0.125f;
  float var = 0.0f;
  #pragma unroll
  for (int vi = 0; vi < VW; ++vi) { float d = v[vi] - mean; var += d * d; }
  var *= (1.0f / 7.0f);          // ddof = 1

  f16* row = shin + (size_t)sr * 320;
  row[c]       = (f16)mean;
  row[128 + c] = (f16)var;
  if (c < 64) {
    float e = 0.0f;
    if (c < 36) {
      int cc = (c < 18) ? c : c - 18;
      int d = cc / 6, f = cc % 6;
      float p = positions[(size_t)sr * 3 + d];
      float ang = p * 3.14159265358979323846f * (float)(1 << f);
      e = (c < 18) ? sinf(ang) : cosf(ang);
    } else if (c < 39) {
      e = positions[(size_t)sr * 3 + (c - 36)];
    }
    row[256 + c] = (f16)e;
  }
}

// ---------------------------------------------------------------------------
// Softmax over 8 views of pooled[...,128], weighted sum of pooled[...,:128].
// ---------------------------------------------------------------------------
__global__ __launch_bounds__(128) void softmax_pool(
    const float* __restrict__ pooled, f16* __restrict__ vin)
{
  int sr = blockIdx.x;
  int r = sr >> 6, s = sr & 63;
  int c = threadIdx.x;
  size_t rows[VW]; float lg[VW]; float mx = -1e30f;
  #pragma unroll
  for (int vi = 0; vi < VW; ++vi) {
    rows[vi] = (size_t)(((r * 8 + vi) * 64) + s) * 132;
    lg[vi] = pooled[rows[vi] + 128];
    mx = fmaxf(mx, lg[vi]);
  }
  float e[VW]; float sum = 0.0f;
  #pragma unroll
  for (int vi = 0; vi < VW; ++vi) { e[vi] = expf(lg[vi] - mx); sum += e[vi]; }
  float inv = 1.0f / sum;
  float acc = 0.0f;
  #pragma unroll
  for (int vi = 0; vi < VW; ++vi) acc += (e[vi] * inv) * pooled[rows[vi] + c];
  vin[(size_t)sr * CH + c] = (f16)acc;
}

// ---------------------------------------------------------------------------
extern "C" void kernel_launch(void* const* d_in, const int* in_sizes, int n_in,
                              void* d_out, int out_size, void* d_ws, size_t ws_size,
                              hipStream_t stream) {
  const float* positions = (const float*)d_in[0];
  const float* w2c       = (const float*)d_in[1];
  const float* fx        = (const float*)d_in[2];
  const float* fy        = (const float*)d_in[3];
  const float* cx        = (const float*)d_in[4];
  const float* cy        = (const float*)d_in[5];
  const float* imgf      = (const float*)d_in[6];
  const float* uvs       = (const float*)d_in[7];
  const float* fp_w1     = (const float*)d_in[8];
  const float* fp_b1     = (const float*)d_in[9];
  const float* fp_w2     = (const float*)d_in[10];
  const float* fp_b2     = (const float*)d_in[11];
  const float* vr_w1     = (const float*)d_in[12];
  const float* vr_b1     = (const float*)d_in[13];
  const float* vr_w2     = (const float*)d_in[14];
  const float* vr_b2     = (const float*)d_in[15];
  const float* vr_w3     = (const float*)d_in[16];
  const float* vr_b3     = (const float*)d_in[17];
  const float* vr_w4     = (const float*)d_in[18];
  const float* vr_b4     = (const float*)d_in[19];
  float* out = (float*)d_out;

  char* wsp = (char*)d_ws;
  size_t off = 0;
  auto alloc = [&](size_t bytes) -> void* {
    void* p = wsp + off;
    off = (off + bytes + 255) & ~(size_t)255;
    return p;
  };

  f16*  WshT  = (f16*)alloc((size_t)256 * 320 * 2);        // shared fp_w1 part, K=320
  f16*  W1aT  = (f16*)alloc((size_t)256 * 128 * 2);        // feats fp_w1 part, K=128
  f16*  W2T   = (f16*)alloc((size_t)144 * 256 * 2);        // fp_w2, N=129 pad 144
  f16*  V1T   = (f16*)alloc((size_t)256 * 128 * 2);
  f16*  V2T   = (f16*)alloc((size_t)256 * 256 * 2);
  f16*  V3T   = (f16*)alloc((size_t)256 * 256 * 2);
  f16*  V4T   = (f16*)alloc((size_t)144 * 256 * 2);        // vr_w4, N=132 pad 144
  f16*  feats = (f16*)alloc((size_t)NROWS_VIEW * 128 * 2);
  f16*  shin  = (f16*)alloc((size_t)NROWS_RAY * 320 * 2);
  float* hs   = (float*)alloc((size_t)NROWS_RAY * 256 * 4);
  f16*  hbuf  = (f16*)alloc((size_t)NROWS_VIEW * 256 * 2);
  float* pooled = (float*)alloc((size_t)NROWS_VIEW * 132 * 4);
  f16*  vin   = (f16*)alloc((size_t)NROWS_RAY * 128 * 2);
  f16*  x1    = (f16*)alloc((size_t)NROWS_RAY * 256 * 2);
  f16*  x2    = (f16*)alloc((size_t)NROWS_RAY * 256 * 2);

  auto prep = [&](const float* W, int ldw, int r0, int nrows,
                  f16* Bt, int ldbt, int k0, int kc, int Npad, int Nsrc) {
    int total = Npad * kc;
    prep_bt<<<(total + 255) / 256, 256, 0, stream>>>(W, ldw, r0, nrows, Bt, ldbt, k0, kc, Npad, Nsrc);
  };
  // fp_w1 rows: [0,128)=feats | [128,256)=mean | [256,384)=var | [384,423)=enc
  prep(fp_w1, 256, 128, 128, WshT, 320,   0, 128, 256, 256);   // mean
  prep(fp_w1, 256, 256, 128, WshT, 320, 128, 128, 256, 256);   // var
  prep(fp_w1, 256, 384,  39, WshT, 320, 256,  64, 256, 256);   // enc (pad 39->64)
  prep(fp_w1, 256,   0, 128, W1aT, 128,   0, 128, 256, 256);   // per-view feats part
  prep(fp_w2, 129,   0, 256, W2T,  256,   0, 256, 144, 129);
  prep(vr_w1, 256,   0, 128, V1T,  128,   0, 128, 256, 256);
  prep(vr_w2, 256,   0, 256, V2T,  256,   0, 256, 256, 256);
  prep(vr_w3, 256,   0, 256, V3T,  256,   0, 256, 256, 256);
  prep(vr_w4, 132,   0, 256, V4T,  256,   0, 256, 144, 132);

  // Projection + bilinear gather (HBM-bound: ~1 GiB of image features)
  project_gather<<<NROWS_VIEW, 128, 0, stream>>>(positions, w2c, fx, fy, cx, cy,
                                                 imgf, uvs, feats);
  // Mean/var over views + NeRF encode
  build_shared<<<NROWS_RAY, 128, 0, stream>>>(feats, positions, shin);

  auto gemm = [&](const f16* A, int lda, const f16* Bt, int ldb,
                  const float* bias, const float* cinit, int cmode,
                  int act, f16* oh, float* of, int ldo, int M, int N, int K) {
    int ntiles = (N + 15) / 16;
    int groups = (ntiles + 3) / 4;
    int totalWaves = (M / 16) * groups;
    int blocks = (totalWaves + 7) / 8;
    wmma_gemm<<<blocks, 256, 0, stream>>>(A, lda, Bt, ldb, bias, cinit, cmode,
                                          act, oh, of, ldo, N, K, groups, totalWaves);
  };

  // hs = [mean|var|enc] @ Wsh + b1          (16384 x 320 x 256, f32 out)
  gemm(shin, 320, WshT, 320, fp_b1, nullptr, 0, 0, nullptr, hs, 256,
       NROWS_RAY, 256, 320);
  // h = relu(feats @ W1a + hs[broadcast])   (131072 x 128 x 256)
  gemm(feats, 128, W1aT, 128, nullptr, hs, 1, 1, hbuf, nullptr, 256,
       NROWS_VIEW, 256, 128);
  // pooled = h @ W2 + b2                    (131072 x 256 x 129)
  gemm(hbuf, 256, W2T, 256, fp_b2, nullptr, 0, 0, nullptr, pooled, 132,
       NROWS_VIEW, 129, 256);
  // softmax over views, weighted pooling -> vin (16384 x 128)
  softmax_pool<<<NROWS_RAY, 128, 0, stream>>>(pooled, vin);
  // vr MLP
  gemm(vin, 128, V1T, 128, vr_b1, nullptr, 0, 1, x1, nullptr, 256,
       NROWS_RAY, 256, 128);
  gemm(x1, 256, V2T, 256, vr_b2, nullptr, 0, 1, x2, nullptr, 256,
       NROWS_RAY, 256, 256);
  gemm(x2, 256, V3T, 256, vr_b3, nullptr, 0, 1, x1, nullptr, 256,
       NROWS_RAY, 256, 256);
  // final: softplus / sigmoid / identity into concatenated d_out blocks
  gemm(x1, 256, V4T, 256, vr_b4, nullptr, 0, 2, nullptr, out, 0,
       NROWS_RAY, 132, 256);
}
